// UltraMemLayerV2_7670811590919
// MI455X (gfx1250) — matrix operand order repaired and checked by reading, not compiled, not run on hardware
//
#include <hip/hip_runtime.h>

#define BS     4096
#define HID    2048
#define KDIM   256
#define KEYN   512
#define NHEAD  2
#define RANK   2
#define KNN    16
#define VDIM   512
#define GCONST 0.10881882041201557f   // 256^-0.4
#define NEG_INF (-3.402823466e38f)

typedef __attribute__((ext_vector_type(16))) __bf16 bf16x16;
typedef __attribute__((ext_vector_type(8)))  float  v8f;

struct __align__(16) U4 { unsigned int a, b, c, d; };
struct __align__(8)  U2 { unsigned int x, y; };
struct __align__(16) F4 { float x, y, z, w; };

union FragBF { bf16x16 v; U4 q[2]; };

__device__ inline unsigned short f2bf(float f) {
    unsigned int u = __float_as_uint(f);
    unsigned int r = u + 0x7FFFu + ((u >> 16) & 1u);   // RNE
    return (unsigned short)(r >> 16);
}
__device__ inline unsigned int pk2(float a, float b) {
    return (unsigned int)f2bf(a) | ((unsigned int)f2bf(b) << 16);
}
__device__ inline float wredsum(float x) {
    #pragma unroll
    for (int o = 16; o >= 1; o >>= 1) x += __shfl_xor(x, o, 32);
    return x;
}
__device__ inline void wargmax(float& v, int& idx) {
    #pragma unroll
    for (int o = 16; o >= 1; o >>= 1) {
        float ov = __shfl_xor(v, o, 32);
        int   oi = __shfl_xor(idx, o, 32);
        if (ov > v || (ov == v && oi < idx)) { v = ov; idx = oi; }
    }
}
// CDNA5 async global->LDS copy of one 16B chunk (tracked by ASYNCcnt).
__device__ inline void async_b128(unsigned lds_addr, const void* gaddr) {
    asm volatile("global_load_async_to_lds_b128 %0, %1, off"
                 :: "v"(lds_addr), "v"((unsigned long long)(size_t)gaddr)
                 : "memory");
}
__device__ inline void wait_async0() {
    asm volatile("s_wait_asynccnt 0x0" ::: "memory");
}
__device__ inline void wait_async4() {   // allow the 4 just-issued copies in flight
    asm volatile("s_wait_asynccnt 0x4" ::: "memory");
}
__device__ inline unsigned lds_off(const void* p) {   // flat LDS addr[31:0] = LDS offset
    return (unsigned)(size_t)p;
}

// ---------------------------------------------------------------------------
// TN GEMM: C[M,N] = A[M,K] * Bt[N,K]^T, bf16 in, fp32 out, bf16 WMMA compute.
// Block: 256 threads (8 waves). Tile 128x128; wave tile 32x64; K-step 32.
// Double-buffered LDS: tile i+1 streams in via GLOBAL_LOAD_ASYNC_TO_LDS_B128
// (ASYNCcnt) while tile i feeds DS_LOAD_B128 fragments -> v_wmma.
// ---------------------------------------------------------------------------
__global__ __launch_bounds__(256) void gemm_tn_bf16(
    const unsigned short* __restrict__ A, const unsigned short* __restrict__ Bt,
    float* __restrict__ C, int N, int K) {
    __shared__ __align__(16) unsigned short As[2][128 * 40];  // pitch 40 bf16 = 80B
    __shared__ __align__(16) unsigned short Bs[2][128 * 40];

    const int tid  = threadIdx.x;
    const int lane = tid & 31;
    const int wave = tid >> 5;
    const int rowBase = blockIdx.y * 128;
    const int colBase = blockIdx.x * 128;
    const int waveM = (wave >> 1) * 32;
    const int waveN = (wave & 1) * 64;
    const int halfSel = lane >> 4;        // 0: lanes 0-15, 1: lanes 16-31
    const int rsel = lane & 15;
    const int c0A = halfSel * 8;          // A frag: K elems 0-7/16-23 or 8-15/24-31
    const int kbB = halfSel * 8;          // B frag: K elems 0-7/16-23 or 8-15/24-31

    // per-thread staging chunks: tile row = e>>2, 16B chunk = e&3 (4 chunks/row)
    const int er0 = tid >> 2,          ec0 = (tid & 3) * 8;
    const int er1 = (tid + 256) >> 2,  ec1 = ec0;

    const unsigned short* Ar0 = A + (size_t)(rowBase + er0) * K + ec0;
    const unsigned short* Ar1 = A + (size_t)(rowBase + er1) * K + ec1;
    const unsigned short* Br0 = Bt + (size_t)(colBase + er0) * K + ec0;
    const unsigned short* Br1 = Bt + (size_t)(colBase + er1) * K + ec1;

    v8f acc[2][4] = {};

    const int nk = K >> 5;
    // prologue: stage tile 0 into buffer 0
    async_b128(lds_off(&As[0][er0 * 40 + ec0]), Ar0);
    async_b128(lds_off(&As[0][er1 * 40 + ec1]), Ar1);
    async_b128(lds_off(&Bs[0][er0 * 40 + ec0]), Br0);
    async_b128(lds_off(&Bs[0][er1 * 40 + ec1]), Br1);

    for (int i = 0; i < nk; ++i) {
        const int cur = i & 1;
        if (i + 1 < nk) {          // stream tile i+1 while computing tile i
            const int nxt = cur ^ 1;
            const int k1 = (i + 1) << 5;
            async_b128(lds_off(&As[nxt][er0 * 40 + ec0]), Ar0 + k1);
            async_b128(lds_off(&As[nxt][er1 * 40 + ec1]), Ar1 + k1);
            async_b128(lds_off(&Bs[nxt][er0 * 40 + ec0]), Br0 + k1);
            async_b128(lds_off(&Bs[nxt][er1 * 40 + ec1]), Br1 + k1);
            wait_async4();         // tile i landed; tile i+1 stays in flight
        } else {
            wait_async0();
        }
        __syncthreads();

        FragBF fa[2], fb[4];
        #pragma unroll
        for (int mi = 0; mi < 2; ++mi) {
            int r = waveM + mi * 16 + rsel;
            fa[mi].q[0] = *(const U4*)&As[cur][r * 40 + c0A];
            fa[mi].q[1] = *(const U4*)&As[cur][r * 40 + c0A + 16];
        }
        #pragma unroll
        for (int ni = 0; ni < 4; ++ni) {
            int r = waveN + ni * 16 + rsel;
            fb[ni].q[0] = *(const U4*)&Bs[cur][r * 40 + kbB];
            fb[ni].q[1] = *(const U4*)&Bs[cur][r * 40 + kbB + 16];
        }
        #pragma unroll
        for (int mi = 0; mi < 2; ++mi)
            #pragma unroll
            for (int ni = 0; ni < 4; ++ni)
                acc[mi][ni] = __builtin_amdgcn_wmma_f32_16x16x32_bf16(
                    false, fa[mi].v, false, fb[ni].v,
                    (short)0, acc[mi][ni], false, false);
        __syncthreads();           // all waves done reading buf 'cur' before refill
    }

    #pragma unroll
    for (int mi = 0; mi < 2; ++mi)
        #pragma unroll
        for (int ni = 0; ni < 4; ++ni) {
            int gc = colBase + waveN + ni * 16 + rsel;
            #pragma unroll
            for (int e = 0; e < 8; ++e) {
                int gr = rowBase + waveM + mi * 16 + e + halfSel * 8;
                C[(size_t)gr * N + gc] = acc[mi][ni][e];
            }
        }
}

// ---------------------------------------------------------------------------
// fp32 -> bf16 bulk conversion: each thread converts 8 contiguous elements.
// ---------------------------------------------------------------------------
__global__ void cvt_bf16_kernel(const float* __restrict__ in,
                                unsigned short* __restrict__ out) {
    size_t i = ((size_t)blockIdx.x * blockDim.x + threadIdx.x) * 8;
    F4 a = *(const F4*)(in + i);
    F4 b = *(const F4*)(in + i + 4);
    U4 o; o.a = pk2(a.x, a.y); o.b = pk2(a.z, a.w);
    o.c = pk2(b.x, b.y); o.d = pk2(b.z, b.w);
    *(U4*)(out + i) = o;
}

// ---------------------------------------------------------------------------
// Query LayerNorm: one wave per (row, half); writes bf16 rows.
// ---------------------------------------------------------------------------
__global__ void qln_kernel(const float* __restrict__ q,
                           unsigned short* __restrict__ qn1,
                           unsigned short* __restrict__ qn2) {
    int b = blockIdx.x, half = blockIdx.y, lane = threadIdx.x;
    const float* row = q + (size_t)b * (2 * KDIM) + half * KDIM + lane * 8;
    F4 v0 = *(const F4*)(row);
    F4 v1 = *(const F4*)(row + 4);
    float s  = v0.x + v0.y + v0.z + v0.w + v1.x + v1.y + v1.z + v1.w;
    float s2 = v0.x*v0.x + v0.y*v0.y + v0.z*v0.z + v0.w*v0.w
             + v1.x*v1.x + v1.y*v1.y + v1.z*v1.z + v1.w*v1.w;
    float mean = wredsum(s) * (1.0f / KDIM);
    float var  = wredsum(s2) * (1.0f / KDIM) - mean * mean;
    float sc   = rsqrtf(var + 1e-5f) * GCONST;
    unsigned short* o = (half ? qn2 : qn1) + (size_t)b * KDIM + lane * 8;
    U4 ob;
    ob.a = pk2((v0.x - mean) * sc, (v0.y - mean) * sc);
    ob.b = pk2((v0.z - mean) * sc, (v0.w - mean) * sc);
    ob.c = pk2((v1.x - mean) * sc, (v1.y - mean) * sc);
    ob.d = pk2((v1.z - mean) * sc, (v1.w - mean) * sc);
    *(U4*)o = ob;
}

// ---------------------------------------------------------------------------
// Key LayerNorm over kdim; writes transposed bf16 weight Wt[col][k],
// col = (h*512+n)*2 + r. One wave per vector. grid: (512, 4(h,side), 2(r))
// ---------------------------------------------------------------------------
__global__ void keynorm_kernel(const float* __restrict__ keys,
                               unsigned short* __restrict__ Wt1,
                               unsigned short* __restrict__ Wt2) {
    int n = blockIdx.x;
    int h = blockIdx.y >> 1, side = blockIdx.y & 1;
    int r = blockIdx.z, lane = threadIdx.x;
    const float* base = keys + ((((size_t)h * 2 + side) * KEYN + n) * KDIM) * RANK + r;
    float vals[8];
    float s = 0.f, sq = 0.f;
    #pragma unroll
    for (int t = 0; t < 8; ++t) {
        float x = base[(size_t)(lane * 8 + t) * RANK];
        vals[t] = x; s += x; sq += x * x;
    }
    float mean = wredsum(s) * (1.0f / KDIM);
    float var  = wredsum(sq) * (1.0f / KDIM) - mean * mean;
    float sc   = rsqrtf(var + 1e-5f) * GCONST;
    int col = (h * KEYN + n) * RANK + r;
    unsigned short* o = (side ? Wt2 : Wt1) + (size_t)col * KDIM + lane * 8;
    U4 ob;
    ob.a = pk2((vals[0] - mean) * sc, (vals[1] - mean) * sc);
    ob.b = pk2((vals[2] - mean) * sc, (vals[3] - mean) * sc);
    ob.c = pk2((vals[4] - mean) * sc, (vals[5] - mean) * sc);
    ob.d = pk2((vals[6] - mean) * sc, (vals[7] - mean) * sc);
    *(U4*)o = ob;
}

// ---------------------------------------------------------------------------
// Top-k / Tucker-score / softmax: one wave per (b, h).
// ---------------------------------------------------------------------------
__global__ void topk_kernel(const float* __restrict__ s1, const float* __restrict__ s2,
                            const float* __restrict__ core0, const float* __restrict__ core1,
                            const float* __restrict__ uvec, const float* __restrict__ vvec,
                            const int* __restrict__ shuf,
                            float* __restrict__ wout, int* __restrict__ vout) {
    int b = blockIdx.x, h = blockIdx.y, lane = threadIdx.x;
    const float* r1 = s1 + (size_t)b * (NHEAD * KEYN * RANK) + h * (KEYN * RANK);
    const float* r2 = s2 + (size_t)b * (NHEAD * KEYN * RANK) + h * (KEYN * RANK);

    float a[16];
    int i1reg = 0, i2reg = 0;

    // --- side 1: a1 = s1 . u ; top-16 of 512 ---
    {
        float u0 = uvec[h * RANK], u1 = uvec[h * RANK + 1];
        #pragma unroll
        for (int i = 0; i < 8; ++i) {
            F4 g = *(const F4*)(r1 + lane * 32 + i * 4);
            a[i * 2]     = g.x * u0 + g.y * u1;
            a[i * 2 + 1] = g.z * u0 + g.w * u1;
        }
        for (int p = 0; p < KNN; ++p) {
            float bv = NEG_INF; int bn = 0x7fffffff;
            #pragma unroll
            for (int t = 0; t < 16; ++t) {
                int n = lane * 16 + t;
                if (a[t] > bv || (a[t] == bv && n < bn)) { bv = a[t]; bn = n; }
            }
            wargmax(bv, bn);
            if (lane == p) i1reg = bn;
            if ((bn >> 4) == lane) a[bn & 15] = NEG_INF;
        }
    }
    // --- side 2 ---
    {
        float u0 = vvec[h * RANK], u1 = vvec[h * RANK + 1];
        #pragma unroll
        for (int i = 0; i < 8; ++i) {
            F4 g = *(const F4*)(r2 + lane * 32 + i * 4);
            a[i * 2]     = g.x * u0 + g.y * u1;
            a[i * 2 + 1] = g.z * u0 + g.w * u1;
        }
        for (int p = 0; p < KNN; ++p) {
            float bv = NEG_INF; int bn = 0x7fffffff;
            #pragma unroll
            for (int t = 0; t < 16; ++t) {
                int n = lane * 16 + t;
                if (a[t] > bv || (a[t] == bv && n < bn)) { bv = a[t]; bn = n; }
            }
            wargmax(bv, bn);
            if (lane == p) i2reg = bn;
            if ((bn >> 4) == lane) a[bn & 15] = NEG_INF;
        }
    }

    // --- gather selected rank-2 scores; fold core C = core0+core1 ---
    float C00 = core0[h * 4 + 0] + core1[h * 4 + 0];
    float C01 = core0[h * 4 + 1] + core1[h * 4 + 1];
    float C10 = core0[h * 4 + 2] + core1[h * 4 + 2];
    float C11 = core0[h * 4 + 3] + core1[h * 4 + 3];
    float t1_0 = 0.f, t1_1 = 0.f, g20 = 0.f, g21 = 0.f;
    if (lane < KNN) {
        float g0 = r1[i1reg * RANK], g1 = r1[i1reg * RANK + 1];
        t1_0 = g0 * C00 + g1 * C10;
        t1_1 = g0 * C01 + g1 * C11;
        g20 = r2[i2reg * RANK];
        g21 = r2[i2reg * RANK + 1];
    }

    // --- 16x16 candidate grid: lane holds 8 entries ---
    int   jj    = lane & 15;
    int   ibase = (lane >> 4) * 8;
    float s20 = __shfl(g20, jj, 32);
    float s21 = __shfl(g21, jj, 32);
    float cv[8];
    #pragma unroll
    for (int t = 0; t < 8; ++t) {
        float u0 = __shfl(t1_0, ibase + t, 32);
        float u1 = __shfl(t1_1, ibase + t, 32);
        cv[t] = u0 * s20 + u1 * s21;
    }

    // --- top-16 of 256 ---
    float tsv = 0.f; int tsf = 0;
    for (int p = 0; p < KNN; ++p) {
        float bv = NEG_INF; int bf = 0x7fffffff;
        #pragma unroll
        for (int t = 0; t < 8; ++t) {
            int f = (ibase + t) * 16 + jj;
            if (cv[t] > bv || (cv[t] == bv && f < bf)) { bv = cv[t]; bf = f; }
        }
        wargmax(bv, bf);
        if (lane == p) { tsv = bv; tsf = bf; }
        int ownerLane = (((bf >> 4) >= 8) ? 16 : 0) | (bf & 15);
        if (lane == ownerLane) cv[(bf >> 4) & 7] = NEG_INF;
    }

    // --- softmax over 16 winners (lanes 0..15; width-16 groups) ---
    float m = tsv;
    #pragma unroll
    for (int o = 8; o >= 1; o >>= 1) m = fmaxf(m, __shfl_xor(m, o, 16));
    float e = __expf(tsv - m);
    float ssum = e;
    #pragma unroll
    for (int o = 8; o >= 1; o >>= 1) ssum += __shfl_xor(ssum, o, 16);
    float wgt = e / ssum;

    int ii  = tsf >> 4;
    int jjf = tsf & 15;
    int key_i = __shfl(i1reg, ii, 32);
    int key_j = __shfl(i2reg, jjf, 32);
    if (lane < KNN) {
        int vid = shuf[key_i * KEYN + key_j];
        size_t o = (size_t)b * (NHEAD * KNN) + h * KNN + lane;
        wout[o] = wgt;
        vout[o] = vid;
    }
}

// ---------------------------------------------------------------------------
// Sparse value gather + weighted combine; writes bf16 out_v (A of final GEMM).
// ---------------------------------------------------------------------------
__global__ void gather_kernel(const float* __restrict__ values,
                              const float* __restrict__ wbuf, const int* __restrict__ vbuf,
                              unsigned short* __restrict__ out_v) {
    __shared__ float wl[NHEAD * KNN];
    __shared__ int   il[NHEAD * KNN];
    int b = blockIdx.x, tid = threadIdx.x;
    if (tid < NHEAD * KNN) {
        wl[tid] = wbuf[(size_t)b * (NHEAD * KNN) + tid];
        il[tid] = vbuf[(size_t)b * (NHEAD * KNN) + tid];
    }
    __syncthreads();
    int d = tid * 4;
    float ax = 0.f, ay = 0.f, az = 0.f, aw = 0.f;
    for (int t = 0; t < NHEAD * KNN; ++t) {
        F4 vv = *(const F4*)(values + (size_t)il[t] * VDIM + d);
        float wv = wl[t];
        ax += wv * vv.x; ay += wv * vv.y; az += wv * vv.z; aw += wv * vv.w;
    }
    U2 p; p.x = pk2(ax, ay); p.y = pk2(az, aw);
    *(U2*)(out_v + (size_t)b * VDIM + d) = p;
}

// ---------------------------------------------------------------------------
extern "C" void kernel_launch(void* const* d_in, const int* in_sizes, int n_in,
                              void* d_out, int out_size, void* d_ws, size_t ws_size,
                              hipStream_t stream) {
    (void)in_sizes; (void)n_in; (void)out_size; (void)ws_size;
    const float* x     = (const float*)d_in[0];
    const float* qw    = (const float*)d_in[1];   // [512, 2048]
    const float* keys  = (const float*)d_in[2];   // [2,2,512,256,2]
    const float* core0 = (const float*)d_in[3];
    const float* core1 = (const float*)d_in[4];
    const float* uvec  = (const float*)d_in[5];
    const float* vvec  = (const float*)d_in[6];
    const float* vals  = (const float*)d_in[7];   // [262144, 512]
    const float* vpw   = (const float*)d_in[8];   // [2048, 512]
    const int*   shuf  = (const int*)d_in[9];
    float* out = (float*)d_out;

    char* ws = (char*)d_ws;
    unsigned short* xb   = (unsigned short*)(ws);                        // 16 MB [4096,2048] bf16
    unsigned short* qwb  = (unsigned short*)(ws + ((size_t)16 << 20));   // 2 MB  [512,2048]
    unsigned short* vpwb = (unsigned short*)(ws + ((size_t)18 << 20));   // 2 MB  [2048,512]
    float*          q    = (float*)         (ws + ((size_t)20 << 20));   // 8 MB  [4096,512]
    unsigned short* qn1b = (unsigned short*)(ws + ((size_t)28 << 20));   // 2 MB  [4096,256]
    unsigned short* qn2b = (unsigned short*)(ws + ((size_t)30 << 20));   // 2 MB
    unsigned short* Wt1b = (unsigned short*)(ws + ((size_t)32 << 20));   // 1 MB  [2048,256]
    unsigned short* Wt2b = (unsigned short*)(ws + ((size_t)33 << 20));   // 1 MB
    float*          s1   = (float*)         (ws + ((size_t)34 << 20));   // 32 MB [4096,2048]
    float*          s2   = (float*)         (ws + ((size_t)66 << 20));   // 32 MB
    float*          wb   = (float*)         (ws + ((size_t)98 << 20));   // 0.5 MB
    int*            vb   = (int*)           (ws + ((size_t)98 << 20) + ((size_t)1 << 19));
    unsigned short* ovb  = (unsigned short*)(ws + ((size_t)99 << 20));   // 4 MB  [4096,512]

    // 0) one-time fp32 -> bf16 conversion of GEMM operands
    cvt_bf16_kernel<<<4096, 256, 0, stream>>>(x, xb);       // 8M elems
    cvt_bf16_kernel<<<512, 256, 0, stream>>>(qw, qwb);      // 1M elems
    cvt_bf16_kernel<<<512, 256, 0, stream>>>(vpw, vpwb);    // 1M elems

    // 1) q = x @ qw^T   (TN: A=[4096,2048], Bt=[512,2048])
    gemm_tn_bf16<<<dim3(512 / 128, BS / 128), 256, 0, stream>>>(xb, qwb, q, 512, HID);
    // 2) per-half query LayerNorm -> bf16
    qln_kernel<<<dim3(BS, 2), 32, 0, stream>>>(q, qn1b, qn2b);
    // 3) key LayerNorm -> transposed bf16 score weights
    keynorm_kernel<<<dim3(KEYN, 4, RANK), 32, 0, stream>>>(keys, Wt1b, Wt2b);
    // 4) s1 = qn1 @ Wt1^T ; s2 = qn2 @ Wt2^T  ([4096,256] x [2048,256]^T)
    gemm_tn_bf16<<<dim3(2048 / 128, BS / 128), 256, 0, stream>>>(qn1b, Wt1b, s1, 2048, KDIM);
    gemm_tn_bf16<<<dim3(2048 / 128, BS / 128), 256, 0, stream>>>(qn2b, Wt2b, s2, 2048, KDIM);
    // 5) candidate selection, Tucker scoring, top-k, softmax, index shuffle
    topk_kernel<<<dim3(BS, NHEAD), 32, 0, stream>>>(s1, s2, core0, core1, uvec, vvec, shuf, wb, vb);
    // 6) sparse gather + weighted combine (HBM-bound: ~268 MB random reads)
    gather_kernel<<<BS, VDIM / 4, 0, stream>>>(vals, wb, vb, ovb);
    // 7) out = out_v @ vpw^T  ([4096,512] x [2048,512]^T)
    gemm_tn_bf16<<<dim3(HID / 128, BS / 128), 256, 0, stream>>>(ovb, vpwb, out, HID, VDIM);
}
